// EGNNBackbone_48593259987072
// MI455X (gfx1250) — matrix operand order repaired and checked by reading, not compiled, not run on hardware
//
#include <hip/hip_runtime.h>

// ---------------- problem constants ----------------
#define NB      32
#define NN      64
#define NNODES  (NB * NN)          // 2048
#define HID     128
#define IN_NF   32
#define NLAYERS 4
#define DEG     63                 // edges per row-node (N-1)
#define NEDGES  (NNODES * DEG)     // 129024
#define NJ      (HID / 16)         // 8 N-tiles of 16
#define TM      2                  // M tiles (of 16 edges) per wave

typedef __attribute__((ext_vector_type(16))) __bf16 bf16x16;
typedef __attribute__((ext_vector_type(8)))  __bf16 bf16x8;
typedef __attribute__((ext_vector_type(8)))  float  f32x8;
typedef __attribute__((ext_vector_type(4)))  float  f32x4;

// ---------------- helpers ----------------
__device__ __forceinline__ f32x8 zero8() {
  f32x8 z;
#pragma unroll
  for (int t = 0; t < 8; ++t) z[t] = 0.0f;
  return z;
}

__device__ __forceinline__ float silu_f(float v) {
  return v * (1.0f / (1.0f + __expf(-v)));
}

// A fragment (16x32 bf16, ISA layout): lane hi-half holds K in [8*hi,8*hi+8) U [16+8*hi,16+8*hi+8)
__device__ __forceinline__ bf16x16 a_frag_f32(const float* __restrict__ rp, int kw, int hi) {
  const int kb = kw + hi * 8;
  f32x4 q0 = *(const f32x4*)(rp + kb);
  f32x4 q1 = *(const f32x4*)(rp + kb + 4);
  f32x4 q2 = *(const f32x4*)(rp + kb + 16);
  f32x4 q3 = *(const f32x4*)(rp + kb + 20);
  bf16x16 a;
#pragma unroll
  for (int t = 0; t < 4; ++t) {
    a[t]      = (__bf16)q0[t];
    a[4 + t]  = (__bf16)q1[t];
    a[8 + t]  = (__bf16)q2[t];
    a[12 + t] = (__bf16)q3[t];
  }
  return a;
}

__device__ __forceinline__ bf16x16 a_frag_bf16(const __bf16* __restrict__ rp, int kw, int hi) {
  const int kb = kw + hi * 8;
  bf16x8 lo = *(const bf16x8*)(rp + kb);
  bf16x8 hh = *(const bf16x8*)(rp + kb + 16);
  bf16x16 a;
#pragma unroll
  for (int t = 0; t < 8; ++t) { a[t] = lo[t]; a[8 + t] = hh[t]; }
  return a;
}

// B fragment from swizzled weights: fragment (w,j) stored as 32 lanes x 32B contiguous (1KB),
// lane l holds n = j*16 + (l&15), k = w*32 + (l>>4)*8 + (t&7) + (t>>3)*16 for element t.
__device__ __forceinline__ bf16x16 b_frag_swz(const __bf16* __restrict__ wt, int frag, int lane) {
  return *(const bf16x16*)(wt + (((size_t)frag * 32 + lane) << 4));
}

__device__ __forceinline__ f32x8 wmma_bf16(bf16x16 a, bf16x16 b, f32x8 c) {
  return __builtin_amdgcn_wmma_f32_16x16x32_bf16(false, a, false, b, (short)0, c,
                                                 false, false);
}

// ------- weight swizzle: fp32 W (L, ldk, N) -> bf16 fragments in WMMA B register order -------
__global__ void egnn_swz_kernel(const float* __restrict__ in, __bf16* __restrict__ out,
                                int L, int K, int N, int ldk) {
  int idx = blockIdx.x * blockDim.x + threadIdx.x;
  int total = L * K * N;
  if (idx >= total) return;
  int l   = idx / (K * N);
  int rem = idx % (K * N);
  int fi     = rem >> 9;          // fragment index (w*njloc + j), 512 elems per fragment
  int within = rem & 511;
  int lane   = within >> 4;
  int t      = within & 15;
  int njloc  = N / 16;
  int w = fi / njloc;
  int j = fi % njloc;
  int k = w * 32 + ((lane >> 4) << 3) + (t & 7) + ((t >> 3) << 4);
  int n = j * 16 + (lane & 15);
  out[idx] = (__bf16)in[((size_t)l * ldk + k) * N + n];
}

// ---------------- embedding + x copy ----------------
__global__ void egnn_emb_kernel(const float* __restrict__ h_in, const float* __restrict__ emb_w,
                                const float* __restrict__ emb_b, const float* __restrict__ x_in,
                                float* __restrict__ h, float* __restrict__ x_a) {
  int idx = blockIdx.x * blockDim.x + threadIdx.x;   // NNODES*HID
  if (idx >= NNODES * HID) return;
  int n = idx >> 7, c = idx & 127;
  float s = emb_b[c];
#pragma unroll
  for (int i = 0; i < IN_NF; ++i) s += h_in[n * IN_NF + i] * emb_w[i * HID + c];
  h[idx] = s;
  if (c < 3) x_a[n * 3 + c] = x_in[n * 3 + c];
}

// ---------------- edge kernel: one wave = 32 edges (2 M-tiles) x 128 cols ----------------
#define WAVES1 4
#define EPW    (16 * TM)           // edges per wave
template <int F32EA>
__global__ void __launch_bounds__(32 * WAVES1)
egnn_edge_kernel(const float* __restrict__ h, const float* __restrict__ x,
                 const int* __restrict__ erow, const int* __restrict__ ecol,
                 const float* __restrict__ ea_f32,
                 const __bf16* __restrict__ w1s, const float* __restrict__ b1,
                 const float* __restrict__ w1_last,
                 const __bf16* __restrict__ w2s, const float* __restrict__ b2,
                 const __bf16* __restrict__ cw1s, const float* __restrict__ cb1,
                 const float* __restrict__ cw2,
                 __bf16* __restrict__ m_out, float* __restrict__ p_out) {
  __shared__ __bf16 s_act[WAVES1][EPW * HID];
  __shared__ float  s_rad[WAVES1][EPW];

  const int lane  = threadIdx.x & 31;
  const int wid   = threadIdx.x >> 5;
  const int tile  = blockIdx.x * WAVES1 + wid;
  const int ebase = tile * EPW;
  if (ebase >= NEDGES) return;
  const int mrow = lane & 15;
  const int hi   = lane >> 4;
  __bf16* act  = s_act[wid];
  float*  radp = s_rad[wid];

  int r[TM], cI[TM];
#pragma unroll
  for (int t = 0; t < TM; ++t) {
    const int eM = ebase + t * 16 + mrow;
    r[t]  = erow[eM];
    cI[t] = ecol[eM];
  }

  __builtin_prefetch(w2s, 0, 0);
  __builtin_prefetch(cw1s, 0, 0);

  // radial per edge (computed by low half-wave, broadcast via LDS)
#pragma unroll
  for (int t = 0; t < TM; ++t) {
    float dx = x[r[t] * 3 + 0] - x[cI[t] * 3 + 0];
    float dy = x[r[t] * 3 + 1] - x[cI[t] * 3 + 1];
    float dz = x[r[t] * 3 + 2] - x[cI[t] * 3 + 2];
    float rad = dx * dx + dy * dy + dz * dz;
    if (hi == 0) radp[t * 16 + mrow] = rad;
  }

  // ---- GEMM1: e_in(32x384) @ W1 (swizzled), radial column as rank-1 below ----
  f32x8 c1[TM][8];
#pragma unroll
  for (int t = 0; t < TM; ++t)
#pragma unroll
    for (int j = 0; j < 8; ++j) c1[t][j] = zero8();

#pragma unroll 1
  for (int seg = 0; seg < 2; ++seg) {
#pragma unroll 1
    for (int w = 0; w < 4; ++w) {
      bf16x16 a[TM];
#pragma unroll
      for (int t = 0; t < TM; ++t) {
        const int node = seg ? cI[t] : r[t];
        a[t] = a_frag_f32(h + (size_t)node * HID, w * 32, hi);
      }
      const int fbase = (seg * 4 + w) * NJ;
#pragma unroll
      for (int j = 0; j < 8; ++j) {
        bf16x16 b = b_frag_swz(w1s, fbase + j, lane);
#pragma unroll
        for (int t = 0; t < TM; ++t) c1[t][j] = wmma_bf16(a[t], b, c1[t][j]);
      }
    }
  }
#pragma unroll 1
  for (int w = 0; w < 4; ++w) {   // edge_attr segment (K 256..383)
    bf16x16 a[TM];
#pragma unroll
    for (int t = 0; t < TM; ++t) {
      const size_t eM = (size_t)(ebase + t * 16 + mrow);
      a[t] = F32EA ? a_frag_f32(ea_f32 + eM * HID, w * 32, hi)
                   : a_frag_bf16(m_out + eM * HID, w * 32, hi);
    }
    const int fbase = (8 + w) * NJ;
#pragma unroll
    for (int j = 0; j < 8; ++j) {
      bf16x16 b = b_frag_swz(w1s, fbase + j, lane);
#pragma unroll
      for (int t = 0; t < TM; ++t) c1[t][j] = wmma_bf16(a[t], b, c1[t][j]);
    }
  }

  // rank-1 radial term + bias + SiLU -> LDS activation tile (C layout: M=v+8*hi+16*t, N=16j+mrow)
#pragma unroll
  for (int t = 0; t < TM; ++t) {
    float radv[8];
#pragma unroll
    for (int v = 0; v < 8; ++v) radv[v] = radp[t * 16 + hi * 8 + v];
#pragma unroll
    for (int j = 0; j < 8; ++j) {
      const int n = j * 16 + mrow;
      const float wl = w1_last[n];
      const float bb = b1[n];
#pragma unroll
      for (int v = 0; v < 8; ++v) {
        float u = c1[t][j][v] + radv[v] * wl + bb;
        act[(t * 16 + v + 8 * hi) * HID + n] = (__bf16)silu_f(u);
      }
    }
  }

  // ---- GEMM2: act(32x128) @ W2 -> m ----
  f32x8 c2[TM][8];
#pragma unroll
  for (int t = 0; t < TM; ++t)
#pragma unroll
    for (int j = 0; j < 8; ++j) c2[t][j] = zero8();
#pragma unroll 1
  for (int w = 0; w < 4; ++w) {
    bf16x16 a[TM];
#pragma unroll
    for (int t = 0; t < TM; ++t)
      a[t] = a_frag_bf16(act + (t * 16 + mrow) * HID, w * 32, hi);
#pragma unroll
    for (int j = 0; j < 8; ++j) {
      bf16x16 b = b_frag_swz(w2s, w * NJ + j, lane);
#pragma unroll
      for (int t = 0; t < TM; ++t) c2[t][j] = wmma_bf16(a[t], b, c2[t][j]);
    }
  }
#pragma unroll
  for (int t = 0; t < TM; ++t)
#pragma unroll
    for (int j = 0; j < 8; ++j) {
      const int n = j * 16 + mrow;
      const float bb = b2[n];
#pragma unroll
      for (int v = 0; v < 8; ++v)
        act[(t * 16 + v + 8 * hi) * HID + n] = (__bf16)silu_f(c2[t][j][v] + bb);
    }

  // copy m tile (LDS bf16) to global
#pragma unroll
  for (int t = 0; t < 2 * TM * 4; ++t) {
    int chunk = t * 32 + lane;   // EPW*HID/8 chunks of 8 bf16
    *(bf16x8*)(m_out + (size_t)ebase * HID + chunk * 8) =
        *(const bf16x8*)(act + chunk * 8);
  }

  // ---- coord GEMM: m @ coord_w1, SiLU, then dot with coord_w2 -> p ----
  f32x8 c3[TM][8];
#pragma unroll
  for (int t = 0; t < TM; ++t)
#pragma unroll
    for (int j = 0; j < 8; ++j) c3[t][j] = zero8();
#pragma unroll 1
  for (int w = 0; w < 4; ++w) {
    bf16x16 a[TM];
#pragma unroll
    for (int t = 0; t < TM; ++t)
      a[t] = a_frag_bf16(act + (t * 16 + mrow) * HID, w * 32, hi);
#pragma unroll
    for (int j = 0; j < 8; ++j) {
      bf16x16 b = b_frag_swz(cw1s, w * NJ + j, lane);
#pragma unroll
      for (int t = 0; t < TM; ++t) c3[t][j] = wmma_bf16(a[t], b, c3[t][j]);
    }
  }
#pragma unroll
  for (int t = 0; t < TM; ++t) {
    float part[8];
#pragma unroll
    for (int v = 0; v < 8; ++v) part[v] = 0.0f;
#pragma unroll
    for (int j = 0; j < 8; ++j) {
      const int n = j * 16 + mrow;
      const float bb = cb1[n];
      const float wv = cw2[n];
#pragma unroll
      for (int v = 0; v < 8; ++v) part[v] += silu_f(c3[t][j][v] + bb) * wv;
    }
    // reduce across the 16 lanes sharing the same half-wave (same set of M rows)
#pragma unroll
    for (int v = 0; v < 8; ++v) {
      float s = part[v];
      s += __shfl_xor(s, 1, 32);
      s += __shfl_xor(s, 2, 32);
      s += __shfl_xor(s, 4, 32);
      s += __shfl_xor(s, 8, 32);
      part[v] = s;
    }
    if (mrow == 0) {
#pragma unroll
      for (int v = 0; v < 8; ++v) p_out[ebase + t * 16 + hi * 8 + v] = part[v];
    }
  }
}

// ---------------- segment sums (edges contiguous per row node: e in [n*63,(n+1)*63)) ----------------
__global__ void egnn_agg_kernel(const __bf16* __restrict__ m, const float* __restrict__ p,
                                const int* __restrict__ ecol, const float* __restrict__ x_cur,
                                float* __restrict__ x_next, float* __restrict__ agg) {
  int idx = blockIdx.x * blockDim.x + threadIdx.x;
  if (idx >= NNODES * HID) return;
  int n = idx >> 7, c = idx & 127;
  const __bf16* mp = m + (size_t)n * DEG * HID + c;
  float s = 0.0f;
#pragma unroll 1
  for (int t = 0; t < DEG; ++t) s += (float)mp[t * HID];
  agg[idx] = s;
  if (c < 3) {
    int eb = n * DEG;
    float xc = x_cur[n * 3 + c];
    float acc = 0.0f;
#pragma unroll 1
    for (int t = 0; t < DEG; ++t) {
      int e = eb + t;
      acc += (xc - x_cur[ecol[e] * 3 + c]) * p[e];
    }
    x_next[n * 3 + c] = xc + acc * (1.0f / (float)DEG);
  }
}

// ---------------- node MLP with residual: one wave = 16 nodes ----------------
#define WAVES2 4
__global__ void __launch_bounds__(32 * WAVES2)
egnn_node_kernel(float* __restrict__ h, const float* __restrict__ agg,
                 const __bf16* __restrict__ nw1s, const float* __restrict__ nb1,
                 const __bf16* __restrict__ nw2s, const float* __restrict__ nb2) {
  __shared__ __bf16 s_act[WAVES2][16 * HID];
  const int lane  = threadIdx.x & 31;
  const int wid   = threadIdx.x >> 5;
  const int tile  = blockIdx.x * WAVES2 + wid;
  const int nbase = tile * 16;
  if (nbase >= NNODES) return;
  const int mrow = lane & 15;
  const int hi   = lane >> 4;
  const int node = nbase + mrow;
  __bf16* act = s_act[wid];

  f32x8 c1[8];
#pragma unroll
  for (int j = 0; j < 8; ++j) c1[j] = zero8();
#pragma unroll 1
  for (int seg = 0; seg < 2; ++seg) {
    const float* rp = (seg ? agg : h) + (size_t)node * HID;
#pragma unroll 1
    for (int w = 0; w < 4; ++w) {
      bf16x16 a = a_frag_f32(rp, w * 32, hi);
      const int fbase = (seg * 4 + w) * NJ;
#pragma unroll
      for (int j = 0; j < 8; ++j)
        c1[j] = wmma_bf16(a, b_frag_swz(nw1s, fbase + j, lane), c1[j]);
    }
  }
#pragma unroll
  for (int j = 0; j < 8; ++j) {
    const int n = j * 16 + mrow;
    const float bb = nb1[n];
#pragma unroll
    for (int v = 0; v < 8; ++v)
      act[(v + 8 * hi) * HID + n] = (__bf16)silu_f(c1[j][v] + bb);
  }

  f32x8 c2[8];
#pragma unroll
  for (int j = 0; j < 8; ++j) c2[j] = zero8();
#pragma unroll 1
  for (int w = 0; w < 4; ++w) {
    bf16x16 a = a_frag_bf16(act + mrow * HID, w * 32, hi);
#pragma unroll
    for (int j = 0; j < 8; ++j)
      c2[j] = wmma_bf16(a, b_frag_swz(nw2s, w * NJ + j, lane), c2[j]);
  }
  // residual add
#pragma unroll
  for (int j = 0; j < 8; ++j) {
    const int n = j * 16 + mrow;
    const float bb = nb2[n];
#pragma unroll
    for (int v = 0; v < 8; ++v) {
      size_t idx = (size_t)(nbase + v + 8 * hi) * HID + n;
      h[idx] = h[idx] + c2[j][v] + bb;
    }
  }
}

__global__ void egnn_copy_kernel(const float* __restrict__ src, float* __restrict__ dst, int n) {
  int idx = blockIdx.x * blockDim.x + threadIdx.x;
  if (idx < n) dst[idx] = src[idx];
}

// ---------------- host launch ----------------
extern "C" void kernel_launch(void* const* d_in, const int* in_sizes, int n_in,
                              void* d_out, int out_size, void* d_ws, size_t ws_size,
                              hipStream_t stream) {
  const float* h_in  = (const float*)d_in[0];
  const float* x_in  = (const float*)d_in[1];
  const int*   erow  = (const int*)d_in[2];
  const int*   ecol  = (const int*)d_in[3];
  const float* ea    = (const float*)d_in[4];
  const float* emb_w = (const float*)d_in[5];
  const float* emb_b = (const float*)d_in[6];
  const float* ew1   = (const float*)d_in[7];
  const float* eb1   = (const float*)d_in[8];
  const float* ew2   = (const float*)d_in[9];
  const float* eb2   = (const float*)d_in[10];
  const float* nw1   = (const float*)d_in[11];
  const float* nb1   = (const float*)d_in[12];
  const float* nw2   = (const float*)d_in[13];
  const float* nb2   = (const float*)d_in[14];
  const float* cw1   = (const float*)d_in[15];
  const float* cb1   = (const float*)d_in[16];
  const float* cw2   = (const float*)d_in[17];

  char* ws = (char*)d_ws;
  auto alloc = [&](size_t bytes) -> char* {
    char* p = ws;
    ws += (bytes + 255) & ~(size_t)255;
    return p;
  };
  float*  h_buf = (float*)alloc((size_t)NNODES * HID * 4);
  float*  x_a   = (float*)alloc((size_t)NNODES * 3 * 4);
  float*  x_b   = (float*)alloc((size_t)NNODES * 3 * 4);
  float*  aggb  = (float*)alloc((size_t)NNODES * HID * 4);
  float*  p_buf = (float*)alloc((size_t)NEDGES * 4);
  __bf16* m_buf = (__bf16*)alloc((size_t)NEDGES * HID * 2);
  __bf16* w1s   = (__bf16*)alloc((size_t)NLAYERS * 384 * HID * 2);
  __bf16* w2s   = (__bf16*)alloc((size_t)NLAYERS * HID * HID * 2);
  __bf16* cw1s  = (__bf16*)alloc((size_t)NLAYERS * HID * HID * 2);
  __bf16* nw1s  = (__bf16*)alloc((size_t)NLAYERS * 256 * HID * 2);
  __bf16* nw2s  = (__bf16*)alloc((size_t)NLAYERS * HID * HID * 2);

  // weight swizzle into WMMA B-fragment order (bf16)
  {
    int t1 = NLAYERS * 384 * HID;
    egnn_swz_kernel<<<(t1 + 255) / 256, 256, 0, stream>>>(ew1, w1s, NLAYERS, 384, HID, 385);
    int t2 = NLAYERS * HID * HID;
    egnn_swz_kernel<<<(t2 + 255) / 256, 256, 0, stream>>>(ew2, w2s, NLAYERS, HID, HID, HID);
    egnn_swz_kernel<<<(t2 + 255) / 256, 256, 0, stream>>>(cw1, cw1s, NLAYERS, HID, HID, HID);
    int t3 = NLAYERS * 256 * HID;
    egnn_swz_kernel<<<(t3 + 255) / 256, 256, 0, stream>>>(nw1, nw1s, NLAYERS, 256, HID, 256);
    egnn_swz_kernel<<<(t2 + 255) / 256, 256, 0, stream>>>(nw2, nw2s, NLAYERS, HID, HID, HID);
  }

  // embedding + x copy
  egnn_emb_kernel<<<(NNODES * HID + 255) / 256, 256, 0, stream>>>(h_in, emb_w, emb_b, x_in,
                                                                  h_buf, x_a);

  float* xc = x_a;
  float* xn = x_b;
  const int edge_blocks = NEDGES / (EPW * WAVES1);               // 1008
  const int node_blocks = (NNODES / 16 + WAVES2 - 1) / WAVES2;   // 32
  for (int i = 0; i < NLAYERS; ++i) {
    const __bf16* w1l  = w1s  + (size_t)i * 384 * HID;
    const __bf16* w2l  = w2s  + (size_t)i * HID * HID;
    const __bf16* cw1l = cw1s + (size_t)i * HID * HID;
    const float*  wl   = ew1 + ((size_t)i * 385 + 384) * HID;   // radial weight row (fp32)
    if (i == 0) {
      egnn_edge_kernel<1><<<edge_blocks, 32 * WAVES1, 0, stream>>>(
          h_buf, xc, erow, ecol, ea, w1l, eb1 + i * HID, wl, w2l, eb2 + i * HID,
          cw1l, cb1 + i * HID, cw2 + (size_t)i * HID, m_buf, p_buf);
    } else {
      egnn_edge_kernel<0><<<edge_blocks, 32 * WAVES1, 0, stream>>>(
          h_buf, xc, erow, ecol, ea, w1l, eb1 + i * HID, wl, w2l, eb2 + i * HID,
          cw1l, cb1 + i * HID, cw2 + (size_t)i * HID, m_buf, p_buf);
    }
    egnn_agg_kernel<<<(NNODES * HID + 255) / 256, 256, 0, stream>>>(m_buf, p_buf, ecol, xc,
                                                                    xn, aggb);
    egnn_node_kernel<<<node_blocks, 32 * WAVES2, 0, stream>>>(
        h_buf, aggb, nw1s + (size_t)i * 256 * HID, nb1 + i * HID,
        nw2s + (size_t)i * HID * HID, nb2 + i * HID);
    float* tmp = xc; xc = xn; xn = tmp;
  }

  egnn_copy_kernel<<<(NNODES * HID + 255) / 256, 256, 0, stream>>>(h_buf, (float*)d_out,
                                                                   NNODES * HID);
}
